// ReactionDiffusionMinGRULayer_24137716203728
// MI455X (gfx1250) — matrix operand order repaired and checked by reading, not compiled, not run on hardware
//
#include <hip/hip_runtime.h>
#include <hip/hip_bf16.h>
#include <math.h>

#define D_MODEL 1024
#define T_LEN   4096
#define TC      128
#define NCHUNK  (T_LEN / TC)   // 32

typedef __attribute__((ext_vector_type(16))) __bf16 v16bf;
typedef __attribute__((ext_vector_type(8)))  float  v8f;

union FragAB {
    v16bf  v;
    float4 f4[2];   // 32 bytes = 16 bf16
};

static __device__ __forceinline__ unsigned short f32_bf16(float f) {
    union { float f; unsigned u; } x; x.f = f;
    unsigned r = x.u + 0x7FFFu + ((x.u >> 16) & 1u);   // round-to-nearest-even
    return (unsigned short)(r >> 16);
}

// ---- CDNA5 async copy: global -> LDS, 16B per lane, tracked by ASYNCcnt ----
static __device__ __forceinline__ void async_g2l_b128(unsigned lds_off,
                                                      unsigned goff_bytes,
                                                      unsigned long long sbase) {
    asm volatile("global_load_async_to_lds_b128 %0, %1, %2"
                 :: "v"(lds_off), "v"(goff_bytes), "s"(sbase) : "memory");
}
static __device__ __forceinline__ void wait_async_le4() {
#if __has_builtin(__builtin_amdgcn_s_wait_asynccnt)
    __builtin_amdgcn_s_wait_asynccnt(4);
#else
    asm volatile("s_wait_asynccnt 0x4" ::: "memory");
#endif
}
static __device__ __forceinline__ void wait_async_0() {
#if __has_builtin(__builtin_amdgcn_s_wait_asynccnt)
    __builtin_amdgcn_s_wait_asynccnt(0);
#else
    asm volatile("s_wait_asynccnt 0x0" ::: "memory");
#endif
}

// ---------------------------------------------------------------------------
// Kernel 1: pack x (fp32) -> bf16, 8 elements / thread, 16B stores
// ---------------------------------------------------------------------------
__global__ void k_prep_x(const float* __restrict__ x,
                         unsigned short* __restrict__ xb, int n8) {
    int i = blockIdx.x * blockDim.x + threadIdx.x;
    if (i >= n8) return;
    const float4* p = (const float4*)x + (size_t)i * 2;
    float4 v0 = p[0], v1 = p[1];
    union { unsigned short s[8]; uint4 u; } o;
    o.s[0] = f32_bf16(v0.x); o.s[1] = f32_bf16(v0.y);
    o.s[2] = f32_bf16(v0.z); o.s[3] = f32_bf16(v0.w);
    o.s[4] = f32_bf16(v1.x); o.s[5] = f32_bf16(v1.y);
    o.s[6] = f32_bf16(v1.z); o.s[7] = f32_bf16(v1.w);
    *((uint4*)xb + i) = o.u;
}

// ---------------------------------------------------------------------------
// Kernel 2: transpose+convert Wz/Wh (row-major [k][n] fp32) -> Wt[g][n][k] bf16
// ---------------------------------------------------------------------------
__global__ void k_prep_w(const float* __restrict__ Wz,
                         const float* __restrict__ Wh,
                         unsigned short* __restrict__ Wt) {
    __shared__ float tile[32][33];
    const int g  = blockIdx.z;
    const float* W = g ? Wh : Wz;
    const int n0 = blockIdx.x * 32, k0 = blockIdx.y * 32;
    const int tx = threadIdx.x, ty = threadIdx.y;
    tile[ty][tx] = W[(size_t)(k0 + ty) * D_MODEL + n0 + tx];
    __syncthreads();
    Wt[(size_t)g * D_MODEL * D_MODEL + (size_t)(n0 + ty) * D_MODEL + k0 + tx] =
        f32_bf16(tile[tx][ty]);
}

// ---------------------------------------------------------------------------
// Kernel 3: dual WMMA GEMM with async-to-LDS double buffering.
//   z = x@Wz+bz, h~ = x@Wh+bh, then a = 1-sigmoid(z) -> a_out,
//   b = sigmoid(z)*h~ -> b_out (d_out, consumed by the scan).
// Block tile 128(M) x 64(N), BK=32, 8 waves each computing 2x2 16x16 tiles
// for BOTH weight matrices. Tiles staged with GLOBAL_LOAD_ASYNC_TO_LDS_B128
// (ASYNCcnt-pipelined, 2 LDS buffers); async loads complete in order, so
// s_wait_asynccnt 4 retires exactly the current buffer's 4 copies.
// ---------------------------------------------------------------------------
#define BM 128
#define BN 64
#define BK 32
#define ASTRIDE 40   // ushorts per row (32 data + pad, keeps 16B alignment)

__global__ __launch_bounds__(256) void k_gemm_zh(
    const unsigned short* __restrict__ xb,   // [BT][1024] bf16
    const unsigned short* __restrict__ Wt,   // [2][1024 n][1024 k] bf16
    const float* __restrict__ bz, const float* __restrict__ bh,
    float* __restrict__ a_out, float* __restrict__ b_out) {

    __shared__ __align__(16) unsigned short As[2][BM * ASTRIDE];
    __shared__ __align__(16) unsigned short Ws[2][2 * BN * ASTRIDE];

    const int tid  = threadIdx.x;
    const int lane = tid & 31;
    const int w    = tid >> 5;        // wave 0..7
    const int wm   = w >> 1;          // 0..3 (M)
    const int wn   = w & 1;           // 0..1 (N)
    const int m0   = blockIdx.x * BM;
    const int n0   = blockIdx.y * BN;
    const int half = lane >> 4;       // K-half selector per ISA layout
    const int l16  = lane & 15;

    // staging coordinates (16B chunk per async op, 4 ops/thread/tile)
    const int qa0 = tid * 2, qa1 = tid * 2 + 1;
    const int ar0 = qa0 >> 2, ac0 = qa0 & 3;
    const int ar1 = qa1 >> 2, ac1 = qa1 & 3;
    const int wg0 = qa0 >> 8, wr0 = (qa0 & 255) >> 2, wc0 = qa0 & 3;
    const int wg1 = qa1 >> 8, wr1 = (qa1 & 255) >> 2, wc1 = qa1 & 3;

    const unsigned asBase = (unsigned)(uintptr_t)&As[0][0];
    const unsigned wsBase = (unsigned)(uintptr_t)&Ws[0][0];
    const unsigned asBufB = BM * ASTRIDE * 2;        // bytes per A buffer
    const unsigned wsBufB = 2 * BN * ASTRIDE * 2;    // bytes per W buffer
    const unsigned long long xbp = (unsigned long long)xb;
    const unsigned long long wtp = (unsigned long long)Wt;

    auto stage = [&](int k0, int buf) {
        async_g2l_b128(asBase + buf * asBufB + (unsigned)(ar0 * ASTRIDE + ac0 * 8) * 2,
                       (unsigned)((m0 + ar0) * D_MODEL + k0 + ac0 * 8) * 2, xbp);
        async_g2l_b128(asBase + buf * asBufB + (unsigned)(ar1 * ASTRIDE + ac1 * 8) * 2,
                       (unsigned)((m0 + ar1) * D_MODEL + k0 + ac1 * 8) * 2, xbp);
        async_g2l_b128(wsBase + buf * wsBufB + (unsigned)((wg0 * BN + wr0) * ASTRIDE + wc0 * 8) * 2,
                       (unsigned)(wg0 * D_MODEL * D_MODEL + (n0 + wr0) * D_MODEL + k0 + wc0 * 8) * 2, wtp);
        async_g2l_b128(wsBase + buf * wsBufB + (unsigned)((wg1 * BN + wr1) * ASTRIDE + wc1 * 8) * 2,
                       (unsigned)(wg1 * D_MODEL * D_MODEL + (n0 + wr1) * D_MODEL + k0 + wc1 * 8) * 2, wtp);
    };

    v8f accZ[2][2], accH[2][2];
    const v8f vzero = {0.f, 0.f, 0.f, 0.f, 0.f, 0.f, 0.f, 0.f};
    for (int i = 0; i < 2; ++i)
        for (int j = 0; j < 2; ++j) { accZ[i][j] = vzero; accH[i][j] = vzero; }

    stage(0, 0);
    int buf = 0;
    for (int k0 = 0; k0 < D_MODEL; k0 += BK) {
        if (k0 + BK < D_MODEL) {
            stage(k0 + BK, buf ^ 1);   // prefetch next tile into other buffer
            wait_async_le4();          // oldest 4 (current buffer) retired
            __builtin_prefetch(xb + (size_t)(m0 + (tid >> 1)) * D_MODEL
                                  + (k0 + 2 * BK) + (tid & 1) * 16, 0, 3);
        } else {
            wait_async_0();
        }
        __syncthreads();               // all waves' current-buffer copies visible

        // ---- fragments per ISA VGPR layouts (wave32)
        FragAB Af[2], Bz[2], Bh[2];
        for (int mt = 0; mt < 2; ++mt) {
            const unsigned short* ap = &As[buf][(wm * 32 + mt * 16 + l16) * ASTRIDE];
            // A 16x32 bf16: half h holds K = {8h..8h+7} then {16+8h..23+8h}
            Af[mt].f4[0] = *(const float4*)(ap + 8 * half);
            Af[mt].f4[1] = *(const float4*)(ap + 16 + 8 * half);
        }
        for (int nt = 0; nt < 2; ++nt) {
            // B 32x16 bf16: half h holds K = {8h..8h+7} then {16+8h..23+8h}
            const unsigned short* pz = &Ws[buf][(0 * BN + wn * 32 + nt * 16 + l16) * ASTRIDE];
            const unsigned short* ph = &Ws[buf][(1 * BN + wn * 32 + nt * 16 + l16) * ASTRIDE];
            Bz[nt].f4[0] = *(const float4*)(pz + 8 * half);
            Bz[nt].f4[1] = *(const float4*)(pz + 16 + 8 * half);
            Bh[nt].f4[0] = *(const float4*)(ph + 8 * half);
            Bh[nt].f4[1] = *(const float4*)(ph + 16 + 8 * half);
        }
        for (int mt = 0; mt < 2; ++mt)
            for (int nt = 0; nt < 2; ++nt) {
                accZ[mt][nt] = __builtin_amdgcn_wmma_f32_16x16x32_bf16(
                    false, Af[mt].v, false, Bz[nt].v, (short)0, accZ[mt][nt], false, false);
                accH[mt][nt] = __builtin_amdgcn_wmma_f32_16x16x32_bf16(
                    false, Af[mt].v, false, Bh[nt].v, (short)0, accH[mt][nt], false, false);
            }
        __syncthreads();               // frag reads done before buffer reuse
        buf ^= 1;
    }

    // ---- fused epilogue: sigmoid gate, write a and b
    for (int mt = 0; mt < 2; ++mt)
        for (int nt = 0; nt < 2; ++nt) {
            int n = n0 + wn * 32 + nt * 16 + l16;
            float bzv = bz[n], bhv = bh[n];
            for (int v = 0; v < 8; ++v) {
                int m = m0 + wm * 32 + mt * 16 + v + 8 * half;   // C/D layout
                float z  = 1.f / (1.f + __expf(-(accZ[mt][nt][v] + bzv)));
                float ht = accH[mt][nt][v] + bhv;
                size_t idx = (size_t)m * D_MODEL + n;
                a_out[idx] = 1.f - z;
                b_out[idx] = z * ht;
            }
        }
}

// ---------------------------------------------------------------------------
// Kernel 4: per-(b,d,chunk) scan aggregates (A = prod a, B = scan with h0=0)
// ---------------------------------------------------------------------------
__global__ void k_scan1(const float* __restrict__ a_arr,
                        const float* __restrict__ b_arr,
                        float* __restrict__ Aagg, float* __restrict__ Bagg) {
    int gid = blockIdx.x * blockDim.x + threadIdx.x;
    int d = gid & (D_MODEL - 1);
    int c = (gid >> 10) & (NCHUNK - 1);
    int b = gid >> 15;
    float A = 1.f, Bv = 0.f;
    size_t base = ((size_t)b * T_LEN + (size_t)c * TC) * D_MODEL + d;
    for (int t = 0; t < TC; ++t) {
        float at = a_arr[base], bt = b_arr[base];
        Bv = at * Bv + bt;
        A *= at;
        base += D_MODEL;
    }
    int i = (b * NCHUNK + c) * D_MODEL + d;
    Aagg[i] = A;
    Bagg[i] = Bv;
}

// ---------------------------------------------------------------------------
// Kernel 5: sequential cross-chunk prefix (32 steps) -> per-chunk carry-in h
// ---------------------------------------------------------------------------
__global__ void k_scan2(const float* __restrict__ Aagg,
                        const float* __restrict__ Bagg,
                        float* __restrict__ Carry) {
    int gid = blockIdx.x * blockDim.x + threadIdx.x;
    int d = gid & (D_MODEL - 1);
    int b = gid >> 10;
    float carry = 0.f;
    for (int c = 0; c < NCHUNK; ++c) {
        int i = (b * NCHUNK + c) * D_MODEL + d;
        Carry[i] = carry;
        carry = Aagg[i] * carry + Bagg[i];
    }
}

// ---------------------------------------------------------------------------
// Kernel 6: replay recurrence per chunk + fused diffusion via LDS row exchange.
// ---------------------------------------------------------------------------
__global__ __launch_bounds__(1024) void k_scan3(
    const float* __restrict__ a_arr, float* __restrict__ hb,
    const float* __restrict__ Carry, const float* __restrict__ log_D) {
    __shared__ float hrow[D_MODEL];
    const int d = threadIdx.x;
    const int c = blockIdx.x & (NCHUNK - 1);
    const int b = blockIdx.x >> 5;
    float h  = Carry[(b * NCHUNK + c) * D_MODEL + d];
    float Dc = __expf(log_D[d]);
    size_t base = ((size_t)b * T_LEN + (size_t)c * TC) * D_MODEL + d;
    for (int t = 0; t < TC; ++t) {
        float at = a_arr[base];
        float bt = hb[base];
        h = at * h + bt;
        hrow[d] = h;
        __syncthreads();
        float hl = hrow[(d + D_MODEL - 1) & (D_MODEL - 1)];
        float hr = hrow[(d + 1) & (D_MODEL - 1)];
        hb[base] = h + Dc * (hl - 2.f * h + hr);
        __syncthreads();
        base += D_MODEL;
    }
}

// ---------------------------------------------------------------------------
extern "C" void kernel_launch(void* const* d_in, const int* in_sizes, int n_in,
                              void* d_out, int out_size, void* d_ws, size_t ws_size,
                              hipStream_t stream) {
    const float* x    = (const float*)d_in[0];
    const float* Wz   = (const float*)d_in[1];
    const float* bz   = (const float*)d_in[2];
    const float* Wh   = (const float*)d_in[3];
    const float* bh   = (const float*)d_in[4];
    const float* logD = (const float*)d_in[5];
    float* out = (float*)d_out;

    const int BT = in_sizes[0] / D_MODEL;   // 16384
    const int Bb = BT / T_LEN;              // 4

    // workspace layout
    char* ws = (char*)d_ws;
    unsigned short* xb = (unsigned short*)ws;                 // BT*1024 bf16
    size_t off = (size_t)BT * D_MODEL * 2;
    unsigned short* Wt = (unsigned short*)(ws + off);         // 2*1024*1024 bf16
    off += 2ull * D_MODEL * D_MODEL * 2;
    float* a_arr = (float*)(ws + off);                        // BT*1024 f32
    off += (size_t)BT * D_MODEL * 4;
    float* Aagg = (float*)(ws + off); off += (size_t)Bb * NCHUNK * D_MODEL * 4;
    float* Bagg = (float*)(ws + off); off += (size_t)Bb * NCHUNK * D_MODEL * 4;
    float* Carry = (float*)(ws + off);

    // 1) pack x to bf16
    int n8 = BT * D_MODEL / 8;
    k_prep_x<<<(n8 + 255) / 256, 256, 0, stream>>>(x, xb, n8);

    // 2) transpose+convert weights
    dim3 bw(32, 32), gw(D_MODEL / 32, D_MODEL / 32, 2);
    k_prep_w<<<gw, bw, 0, stream>>>(Wz, Wh, Wt);

    // 3) dual WMMA GEMM (async-to-LDS pipelined) + gate epilogue
    dim3 gg(BT / BM, D_MODEL / BN);
    k_gemm_zh<<<gg, 256, 0, stream>>>(xb, Wt, bz, bh, a_arr, out);

    // 4-6) chunked parallel scan + fused diffusion
    k_scan1<<<(Bb * NCHUNK * D_MODEL) / 256, 256, 0, stream>>>(a_arr, out, Aagg, Bagg);
    k_scan2<<<(Bb * D_MODEL) / 256, 256, 0, stream>>>(Aagg, Bagg, Carry);
    k_scan3<<<Bb * NCHUNK, 1024, 0, stream>>>(a_arr, out, Carry, logD);
}
